// MultiHeadAttention_32512902430830
// MI455X (gfx1250) — matrix operand order repaired and checked by reading, not compiled, not run on hardware
//
#include <hip/hip_runtime.h>
#include <hip/hip_bf16.h>

// ---------------------------------------------------------------------------
// Multi-head attention for MI455X (gfx1250)
//   - all GEMMs on v_wmma_f32_16x16x32_f16 (fp32 accumulate)
//   - 2x2 register blocking in projection GEMMs (A/B fragment reuse)
//   - fused scores+softmax+context kernel: 16x1024 fp32 score slab in LDS,
//     wave-level shuffle softmax, single nontemporal streamed write of attns
// B=4, S=1024, D=1024, H=16, dk=64
// ---------------------------------------------------------------------------

typedef __attribute__((ext_vector_type(16))) _Float16 v16h;
typedef __attribute__((ext_vector_type(8)))  _Float16 v8h;
typedef __attribute__((ext_vector_type(4)))  _Float16 v4h;
typedef __attribute__((ext_vector_type(8)))  float    v8f;
typedef __attribute__((ext_vector_type(4)))  float    v4f;

#define BX 4
#define SX 1024
#define DX 1024
#define HX 16
#define DKX 64
#define MROWS (BX * SX)   // 4096 rows for [B*S, D] GEMMs
#define STRIP 16          // query rows per fused-attention block

union V16U { v16h v; v8h h[2]; };

__device__ __forceinline__ v8f wmma32(v16h a, v16h b, v8f c) {
  return __builtin_amdgcn_wmma_f32_16x16x32_f16(false, a, false, b, (short)0, c,
                                                false, false);
}

// A-fragment (16x32 f16), row-major source, row stride ldk (elements)
__device__ __forceinline__ v16h load_a_f16(const _Float16* base, int ldk, int m,
                                           int k0, int hid) {
  const _Float16* p = base + (size_t)m * ldk + k0;
  V16U u;
  u.h[0] = *(const v8h*)(p + hid * 16);
  u.h[1] = *(const v8h*)(p + hid * 16 + 8);
  return u.v;
}

// B-fragment (32x16 f16): 16 contiguous halves along K of "column" n
__device__ __forceinline__ v16h load_b_f16(const _Float16* p) {
  V16U u;
  u.h[0] = *(const v8h*)(p);
  u.h[1] = *(const v8h*)(p + 8);
  return u.v;
}

__device__ __forceinline__ v8h cvt8_f32_to_f16(const float* p) {
  v4f x0 = *(const v4f*)(p);
  v4f x1 = *(const v4f*)(p + 4);
  v8h r;
#pragma unroll
  for (int i = 0; i < 4; ++i) { r[i] = (_Float16)x0[i]; r[4 + i] = (_Float16)x1[i]; }
  return r;
}

// ---------------------------------------------------------------------------
// K0: fp32 -> f16 convert, 4 elements/thread (n must be a multiple of 1024)
// ---------------------------------------------------------------------------
__global__ __launch_bounds__(256) void cvt_f32_f16_kernel(const float* __restrict__ in,
                                                          _Float16* __restrict__ out) {
  const int i = (blockIdx.x * 256 + threadIdx.x) * 4;
  v4f x = *(const v4f*)(in + i);
  v4h r;
#pragma unroll
  for (int j = 0; j < 4; ++j) r[j] = (_Float16)x[j];
  *(v4h*)(out + i) = r;
}

// ---------------------------------------------------------------------------
// K1: projection GEMM  out[m,n] = sum_k A[m,k] * W[n,k] + bias[n]
// 2x2 register blocking: each wave computes a 32x32 output block.
//   mode 0: store head-major  out[((b*H+h)*S + s)*dk + d]         (Q, K)
//   mode 1: store transposed  out[((b*H+h)*dk + d)*S + s]         (V)
// M=4096, N=1024, K=1024; wave tiles = 128x32 = 4096; grid = 512 x 256thr
// ---------------------------------------------------------------------------
__global__ __launch_bounds__(256) void proj_gemm_kernel(
    const _Float16* __restrict__ A, const _Float16* __restrict__ W,
    const float* __restrict__ bias, _Float16* __restrict__ out, int mode) {
  const int w = threadIdx.x >> 5;
  const int lane = threadIdx.x & 31;
  const int sub = lane & 15;
  const int hid = lane >> 4;
  const int wg = blockIdx.x * 8 + w;
  const int nt = wg & 31;        // 32 blocks along N=1024
  const int mt = wg >> 5;        // 128 blocks along M=4096
  const int m0 = mt * 32, n0 = nt * 32;

  v8f acc[2][2] = {};
  const _Float16* a0p = A + (size_t)(m0 + sub) * DX;
  const _Float16* a1p = A + (size_t)(m0 + 16 + sub) * DX;
  const _Float16* b0p = W + (size_t)(n0 + sub) * DX;
  const _Float16* b1p = W + (size_t)(n0 + 16 + sub) * DX;
#pragma unroll 2
  for (int k0 = 0; k0 < DX; k0 += 32) {
    V16U a0, a1, b0, b1;
    a0.h[0] = *(const v8h*)(a0p + k0 + hid * 16);
    a0.h[1] = *(const v8h*)(a0p + k0 + hid * 16 + 8);
    a1.h[0] = *(const v8h*)(a1p + k0 + hid * 16);
    a1.h[1] = *(const v8h*)(a1p + k0 + hid * 16 + 8);
    b0.h[0] = *(const v8h*)(b0p + k0 + hid * 16);
    b0.h[1] = *(const v8h*)(b0p + k0 + hid * 16 + 8);
    b1.h[0] = *(const v8h*)(b1p + k0 + hid * 16);
    b1.h[1] = *(const v8h*)(b1p + k0 + hid * 16 + 8);
    // prefetch the streaming A operand ~512B ahead (global_prefetch_b8)
    __builtin_prefetch(a0p + k0 + 256, 0, 3);
    __builtin_prefetch(a1p + k0 + 256, 0, 3);
    acc[0][0] = wmma32(a0.v, b0.v, acc[0][0]);
    acc[0][1] = wmma32(a0.v, b1.v, acc[0][1]);
    acc[1][0] = wmma32(a1.v, b0.v, acc[1][0]);
    acc[1][1] = wmma32(a1.v, b1.v, acc[1][1]);
  }

  const int bidx = m0 >> 10;  // 32-row block never straddles a batch (1024%32==0)
#pragma unroll
  for (int ni = 0; ni < 2; ++ni) {
    const int n = n0 + ni * 16 + sub;
    const float bv = bias[n];
    const int h = n >> 6, d = n & 63;
#pragma unroll
    for (int mi = 0; mi < 2; ++mi) {
      if (mode == 0) {
#pragma unroll
        for (int r = 0; r < 8; ++r) {
          const int s = ((m0 + mi * 16) & (SX - 1)) + hid * 8 + r;
          out[(((size_t)(bidx * HX + h) * SX) + s) * DKX + d] =
              (_Float16)(acc[mi][ni][r] + bv);
        }
      } else {
        v8h vals;
#pragma unroll
        for (int r = 0; r < 8; ++r) vals[r] = (_Float16)(acc[mi][ni][r] + bv);
        const int s0 = ((m0 + mi * 16) & (SX - 1)) + hid * 8;
        *(v8h*)(out + ((size_t)(bidx * HX + h) * DKX + d) * SX + s0) = vals;
      }
    }
  }
}

// ---------------------------------------------------------------------------
// K2: fused attention for one (b,h) x 16-query-row strip.
//  stage 1: scores(16x1024) = Q.K^T/8 (masked) -> LDS (fp32, 64KB)
//  stage 2: per-row softmax (wave shuffle reductions), normalized probs back
//           to LDS and streamed to global attns with nontemporal stores
//  stage 3: context(16x64) = P @ V, K-dim split across wave pairs, LDS combine
// grid = 64 bh * 64 strips = 4096 blocks x 256 threads
// ---------------------------------------------------------------------------
__global__ __launch_bounds__(256) void attn_fused_kernel(
    const _Float16* __restrict__ Q, const _Float16* __restrict__ Kp,
    const _Float16* __restrict__ Vt, const unsigned char* __restrict__ mask,
    float* __restrict__ attns, _Float16* __restrict__ ctx) {
  __shared__ float sc[STRIP * SX];  // 64 KB score slab

  const int w = threadIdx.x >> 5;
  const int lane = threadIdx.x & 31;
  const int sub = lane & 15;
  const int hid = lane >> 4;
  const int bh = blockIdx.x >> 6;      // 64 strips per (b,h)
  const int strip = blockIdx.x & 63;
  const int m0 = strip * STRIP;        // query-row base within this bh
  const int bidx = bh >> 4;
  const int h = bh & 15;

  // ---- stage 1: scores into LDS ----
  const _Float16* Qb = Q + (size_t)bh * SX * DKX;
  const _Float16* Kb = Kp + (size_t)bh * SX * DKX;
  v16h afrag[2];
#pragma unroll
  for (int ks = 0; ks < 2; ++ks)
    afrag[ks] = load_a_f16(Qb, DKX, m0 + sub, ks * 32, hid);

#pragma unroll 2
  for (int nc = 0; nc < 8; ++nc) {   // wave covers columns [w*128, w*128+128)
    const int n0 = w * 128 + nc * 16;
    const _Float16* kcol = Kb + (size_t)(n0 + sub) * DKX + hid * 16;
    v16h b0 = load_b_f16(kcol);
    v16h b1 = load_b_f16(kcol + 32);
    v8f acc = {};
    acc = wmma32(afrag[0], b0, acc);
    acc = wmma32(afrag[1], b1, acc);
    const int tcol = n0 + sub;
#pragma unroll
    for (int r = 0; r < 8; ++r) {
      const int ml = hid * 8 + r;
      float sv = acc[r] * 0.125f;  // 1/sqrt(64)
      if (mask[((size_t)bidx * SX + m0 + ml) * SX + tcol]) sv = -__builtin_inff();
      sc[ml * SX + tcol] = sv;
    }
  }
  __syncthreads();

  // ---- stage 2: softmax, 2 rows per wave, shuffle reductions ----
#pragma unroll
  for (int rr = 0; rr < 2; ++rr) {
    const int row = w * 2 + rr;
    float* srow = sc + row * SX;
    float mx = -__builtin_inff();
    for (int i = 0; i < 32; ++i) mx = fmaxf(mx, srow[lane + i * 32]);
#pragma unroll
    for (int off = 16; off > 0; off >>= 1) mx = fmaxf(mx, __shfl_xor(mx, off));
    float sum = 0.f;
    for (int i = 0; i < 32; ++i) {
      const float e = __expf(srow[lane + i * 32] - mx);
      srow[lane + i * 32] = e;
      sum += e;
    }
#pragma unroll
    for (int off = 16; off > 0; off >>= 1) sum += __shfl_xor(sum, off);
    const float inv = 1.0f / sum;
    float* arow = attns + ((size_t)bh * SX + m0 + row) * SX;
    for (int i = 0; i < 32; ++i) {
      const float p = srow[lane + i * 32] * inv;
      srow[lane + i * 32] = p;
      __builtin_nontemporal_store(p, arow + lane + i * 32);  // stream attns out
    }
  }
  __syncthreads();

  // ---- stage 3: context = P(16x1024) @ V(1024x64) ----
  // wave -> (nt = w&3, khalf = w>>2); each accumulates over 512 of K
  const int nt = w & 3;
  const int khalf = w >> 2;
  v8f cacc = {};
  const _Float16* vcol = Vt + ((size_t)bh * DKX + nt * 16 + sub) * SX + hid * 16;
#pragma unroll 2
  for (int ki = 0; ki < 16; ++ki) {
    const int k0 = khalf * 512 + ki * 32;
    const float* rowp = sc + sub * SX + k0;
    V16U ua;
    ua.h[0] = cvt8_f32_to_f16(rowp + hid * 16);
    ua.h[1] = cvt8_f32_to_f16(rowp + hid * 16 + 8);
    v16h b = load_b_f16(vcol + k0);
    cacc = wmma32(ua.v, b, cacc);
  }
  __syncthreads();
  // combine the two K-halves through (now free) LDS
  if (khalf == 1) {
#pragma unroll
    for (int r = 0; r < 8; ++r) sc[nt * 256 + lane * 8 + r] = cacc[r];
  }
  __syncthreads();
  if (khalf == 0) {
#pragma unroll
    for (int r = 0; r < 8; ++r) {
      const float val = cacc[r] + sc[nt * 256 + lane * 8 + r];
      const int m = m0 + hid * 8 + r;  // s within batch
      ctx[((size_t)bidx * SX + m) * DX + h * DKX + nt * 16 + sub] = (_Float16)val;
    }
  }
}

// ---------------------------------------------------------------------------
// K3a: output projection + residual, 2x2 blocked:
//   x[m,n] = ctx[m,:] @ Wo[n,:] + bo[n] + v[m,n]
// ---------------------------------------------------------------------------
__global__ __launch_bounds__(256) void outproj_kernel(
    const _Float16* __restrict__ ctx, const _Float16* __restrict__ Wo,
    const float* __restrict__ bo, const float* __restrict__ vres,
    float* __restrict__ xbuf) {
  const int w = threadIdx.x >> 5;
  const int lane = threadIdx.x & 31;
  const int sub = lane & 15;
  const int hid = lane >> 4;
  const int wg = blockIdx.x * 8 + w;
  const int nt = wg & 31;
  const int mt = wg >> 5;
  const int m0 = mt * 32, n0 = nt * 32;

  v8f acc[2][2] = {};
  const _Float16* a0p = ctx + (size_t)(m0 + sub) * DX;
  const _Float16* a1p = ctx + (size_t)(m0 + 16 + sub) * DX;
  const _Float16* b0p = Wo + (size_t)(n0 + sub) * DX;
  const _Float16* b1p = Wo + (size_t)(n0 + 16 + sub) * DX;
#pragma unroll 2
  for (int k0 = 0; k0 < DX; k0 += 32) {
    V16U a0, a1, b0, b1;
    a0.h[0] = *(const v8h*)(a0p + k0 + hid * 16);
    a0.h[1] = *(const v8h*)(a0p + k0 + hid * 16 + 8);
    a1.h[0] = *(const v8h*)(a1p + k0 + hid * 16);
    a1.h[1] = *(const v8h*)(a1p + k0 + hid * 16 + 8);
    b0.h[0] = *(const v8h*)(b0p + k0 + hid * 16);
    b0.h[1] = *(const v8h*)(b0p + k0 + hid * 16 + 8);
    b1.h[0] = *(const v8h*)(b1p + k0 + hid * 16);
    b1.h[1] = *(const v8h*)(b1p + k0 + hid * 16 + 8);
    __builtin_prefetch(a0p + k0 + 256, 0, 3);
    __builtin_prefetch(a1p + k0 + 256, 0, 3);
    acc[0][0] = wmma32(a0.v, b0.v, acc[0][0]);
    acc[0][1] = wmma32(a0.v, b1.v, acc[0][1]);
    acc[1][0] = wmma32(a1.v, b0.v, acc[1][0]);
    acc[1][1] = wmma32(a1.v, b1.v, acc[1][1]);
  }

#pragma unroll
  for (int ni = 0; ni < 2; ++ni) {
    const int n = n0 + ni * 16 + sub;
    const float bv = bo[n];
#pragma unroll
    for (int mi = 0; mi < 2; ++mi) {
#pragma unroll
      for (int r = 0; r < 8; ++r) {
        const size_t idx = (size_t)(m0 + mi * 16 + hid * 8 + r) * DX + n;
        xbuf[idx] = acc[mi][ni][r] + bv + vres[idx];
      }
    }
  }
}

// ---------------------------------------------------------------------------
// K3b: LayerNorm over last dim (1024); one block per row
// ---------------------------------------------------------------------------
__global__ __launch_bounds__(256) void layernorm_kernel(
    const float* __restrict__ x, const float* __restrict__ g,
    const float* __restrict__ b, float* __restrict__ out) {
  __shared__ float red[256];
  const int tid = threadIdx.x;
  const float* row = x + (size_t)blockIdx.x * DX;

  float v[4];
  float sum = 0.f;
#pragma unroll
  for (int i = 0; i < 4; ++i) { v[i] = row[tid + i * 256]; sum += v[i]; }
  red[tid] = sum; __syncthreads();
  for (int s = 128; s > 0; s >>= 1) {
    if (tid < s) red[tid] += red[tid + s];
    __syncthreads();
  }
  const float mu = red[0] * (1.0f / DX); __syncthreads();

  float vs = 0.f;
#pragma unroll
  for (int i = 0; i < 4; ++i) { float d = v[i] - mu; vs += d * d; }
  red[tid] = vs; __syncthreads();
  for (int s = 128; s > 0; s >>= 1) {
    if (tid < s) red[tid] += red[tid + s];
    __syncthreads();
  }
  const float rstd = rsqrtf(red[0] * (1.0f / DX) + 1e-6f);

  float* orow = out + (size_t)blockIdx.x * DX;
#pragma unroll
  for (int i = 0; i < 4; ++i) {
    const int c = tid + i * 256;
    orow[c] = (v[i] - mu) * rstd * g[c] + b[c];
  }
}

// ---------------------------------------------------------------------------
extern "C" void kernel_launch(void* const* d_in, const int* in_sizes, int n_in,
                              void* d_out, int out_size, void* d_ws, size_t ws_size,
                              hipStream_t stream) {
  const float* q    = (const float*)d_in[0];
  const float* k    = (const float*)d_in[1];
  const float* v    = (const float*)d_in[2];
  const unsigned char* mask = (const unsigned char*)d_in[3];
  const float* Wq   = (const float*)d_in[4];
  const float* bq   = (const float*)d_in[5];
  const float* Wk   = (const float*)d_in[6];
  const float* bk   = (const float*)d_in[7];
  const float* Wv   = (const float*)d_in[8];
  const float* bv   = (const float*)d_in[9];
  const float* Wo   = (const float*)d_in[10];
  const float* bo   = (const float*)d_in[11];
  const float* ln_g = (const float*)d_in[12];
  const float* ln_b = (const float*)d_in[13];

  float* norm_out = (float*)d_out;                              // [B,S,D] = 4M floats
  float* attns    = norm_out + (size_t)BX * SX * DX;            // [B,H,S,S] = 64M floats

  // workspace layout (bytes)
  char* ws = (char*)d_ws;
  const size_t MB = 1024 * 1024;
  _Float16* qb  = (_Float16*)(ws + 0);        // 8 MB  (reused later as xbuf)
  _Float16* kb  = (_Float16*)(ws + 8 * MB);   // 8 MB
  _Float16* vb  = (_Float16*)(ws + 16 * MB);  // 8 MB
  _Float16* Wqb = (_Float16*)(ws + 24 * MB);  // 2 MB
  _Float16* Wkb = (_Float16*)(ws + 26 * MB);  // 2 MB
  _Float16* Wvb = (_Float16*)(ws + 28 * MB);  // 2 MB
  _Float16* Wob = (_Float16*)(ws + 30 * MB);  // 2 MB
  _Float16* Qh  = (_Float16*)(ws + 32 * MB);  // 8 MB  [B,H,S,dk]
  _Float16* Kh  = (_Float16*)(ws + 40 * MB);  // 8 MB  [B,H,S,dk]
  _Float16* Vt  = (_Float16*)(ws + 48 * MB);  // 8 MB  [B,H,dk,S]
  _Float16* ctx = (_Float16*)(ws + 56 * MB);  // 8 MB  [B,S,D]
  float*    xbuf = (float*)(ws + 0);          // 16 MB, reuses qb+kb region

  const int nAct = BX * SX * DX;   // 4M
  const int nW   = DX * DX;        // 1M

  // K0: converts (4 elems/thread)
  cvt_f32_f16_kernel<<<nAct / 1024, 256, 0, stream>>>(q, qb);
  cvt_f32_f16_kernel<<<nAct / 1024, 256, 0, stream>>>(k, kb);
  cvt_f32_f16_kernel<<<nAct / 1024, 256, 0, stream>>>(v, vb);
  cvt_f32_f16_kernel<<<nW / 1024, 256, 0, stream>>>(Wq, Wqb);
  cvt_f32_f16_kernel<<<nW / 1024, 256, 0, stream>>>(Wk, Wkb);
  cvt_f32_f16_kernel<<<nW / 1024, 256, 0, stream>>>(Wv, Wvb);
  cvt_f32_f16_kernel<<<nW / 1024, 256, 0, stream>>>(Wo, Wob);

  // K1: QKV projections, 2x2 blocked (512 blocks = 4096 wave-tiles of 32x32)
  proj_gemm_kernel<<<512, 256, 0, stream>>>(qb, Wqb, bq, Qh, 0);
  proj_gemm_kernel<<<512, 256, 0, stream>>>(kb, Wkb, bk, Kh, 0);
  proj_gemm_kernel<<<512, 256, 0, stream>>>(vb, Wvb, bv, Vt, 1);

  // K2: fused scores + softmax + attns stream-out + context
  attn_fused_kernel<<<64 * (SX / STRIP), 256, 0, stream>>>(Qh, Kh, Vt, mask,
                                                           attns, ctx);

  // K3a: output projection + residual
  outproj_kernel<<<512, 256, 0, stream>>>(ctx, Wob, bo, v, xbuf);

  // K3b: LayerNorm
  layernorm_kernel<<<MROWS, 256, 0, stream>>>(xbuf, ln_g, ln_b, norm_out);
}
